// MutiHeadAttention_41446434406810
// MI455X (gfx1250) — matrix operand order repaired and checked by reading, not compiled, not run on hardware
//
#include <hip/hip_runtime.h>
#include <hip/hip_bf16.h>

typedef __attribute__((ext_vector_type(16))) _Float16 v16h;
typedef __attribute__((ext_vector_type(8)))  _Float16 v8h;
typedef __attribute__((ext_vector_type(8)))  float    v8f;

#define BATCH 4
#define SEQ   2048
#define DM    512
#define NHEAD 8
#define HDIM  64

union V16U { v16h v; v8h h[2]; };

// ---------------- elementwise converts ----------------
__global__ void cvt_f32_f16(const float* __restrict__ in, _Float16* __restrict__ out, int n) {
    int i = blockIdx.x * blockDim.x + threadIdx.x;
    if (i < n) out[i] = (_Float16)in[i];
}

// W [DM][DM] row-major f32 -> WT[n][k] = W[k][n] f16
__global__ void cvt_transpose_f32_f16(const float* __restrict__ W, _Float16* __restrict__ WT) {
    int i = blockIdx.x * blockDim.x + threadIdx.x;   // i = n*DM + k
    int n = i >> 9, k = i & 511;
    WT[i] = (_Float16)W[k * DM + n];
}

// ---------------- WMMA GEMM: C = A[M,K] * BT[N,K]^T + bias ----------------
// Each wave computes a 16x64 output strip (4 n-tiles) so one A fragment feeds
// 4 WMMAs per 32-K step. Block = 8 waves = 16 x 512 of C.
// grid = (M/16, N/512).
// mode 0: f16 out row-major [M,N]
// mode 1: f16 out in vpT layout [b][N][SEQ]  (t = m, b = t/SEQ)
// mode 2: f32 out row-major [M,N]
__global__ __launch_bounds__(256)
void gemm_wmma(const _Float16* __restrict__ A, const _Float16* __restrict__ BT,
               const float* __restrict__ bias, void* __restrict__ out,
               int M, int N, int K, int mode) {
    const int wave = threadIdx.x >> 5;
    const int lane = threadIdx.x & 31;
    const int half = lane >> 4;
    const int mr   = lane & 15;
    const int m0 = blockIdx.x * 16;
    const int n0 = (blockIdx.y * 8 + wave) * 64;
    const int m  = m0 + mr;                      // A row this lane feeds
    const _Float16* Arow = A + (size_t)m * K;
    const _Float16* Brow0 = BT + (size_t)(n0 +  0 + mr) * K;
    const _Float16* Brow1 = BT + (size_t)(n0 + 16 + mr) * K;
    const _Float16* Brow2 = BT + (size_t)(n0 + 32 + mr) * K;
    const _Float16* Brow3 = BT + (size_t)(n0 + 48 + mr) * K;

    v8f acc[4] = {};
    for (int kb = 0; kb < K; kb += 32) {
        V16U a;
        // A fragment: halfword j -> kk = kb + 16*half + j (contiguous)
        a.h[0] = *(const v8h*)(Arow + kb + 16 * half);
        a.h[1] = *(const v8h*)(Arow + kb + 8 + 16 * half);
        // B fragments: halfword j -> kk = kb + 16*half + j (contiguous)
        V16U b0, b1, b2, b3;
        b0.v = *(const v16h*)(Brow0 + kb + 16 * half);
        b1.v = *(const v16h*)(Brow1 + kb + 16 * half);
        b2.v = *(const v16h*)(Brow2 + kb + 16 * half);
        b3.v = *(const v16h*)(Brow3 + kb + 16 * half);
        acc[0] = __builtin_amdgcn_wmma_f32_16x16x32_f16(false, a.v, false, b0.v, (short)0, acc[0], false, false);
        acc[1] = __builtin_amdgcn_wmma_f32_16x16x32_f16(false, a.v, false, b1.v, (short)0, acc[1], false, false);
        acc[2] = __builtin_amdgcn_wmma_f32_16x16x32_f16(false, a.v, false, b2.v, (short)0, acc[2], false, false);
        acc[3] = __builtin_amdgcn_wmma_f32_16x16x32_f16(false, a.v, false, b3.v, (short)0, acc[3], false, false);
    }

    #pragma unroll
    for (int nt = 0; nt < 4; ++nt) {
        const int n = n0 + nt * 16 + mr;         // C column this lane owns
        const float bb = bias[n];
        if (mode == 2) {
            float* o = (float*)out;
            #pragma unroll
            for (int r = 0; r < 8; ++r)
                o[(size_t)(m0 + r + 8 * half) * N + n] = acc[nt][r] + bb;
        } else if (mode == 1) {
            _Float16* o = (_Float16*)out;
            #pragma unroll
            for (int r = 0; r < 8; ++r) {
                int t = m0 + r + 8 * half;
                int bi = t >> 11;                // t / SEQ
                o[((size_t)bi * N + n) * SEQ + (t & (SEQ - 1))] = (_Float16)(acc[nt][r] + bb);
            }
        } else {
            _Float16* o = (_Float16*)out;
            #pragma unroll
            for (int r = 0; r < 8; ++r)
                o[(size_t)(m0 + r + 8 * half) * N + n] = (_Float16)(acc[nt][r] + bb);
        }
    }
}

// ---------------- fused attention: scores + softmax-over-heads + attn*vp ----------------
// qp/kp: [B*SEQ][DM] f16 row-major.  vpT: [B][DM][SEQ] f16.  ctx out: [B*SEQ][DM] f16.
// Softmax is over the HEAD axis (reference quirk) => purely local per (b,q,k) element.
__global__ __launch_bounds__(256)
void attn_kernel(const _Float16* __restrict__ qp, const _Float16* __restrict__ kp,
                 const _Float16* __restrict__ vpT, _Float16* __restrict__ ctx) {
    __shared__ float S[NHEAD][16][32];            // 16 KB score/attn exchange

    const int h    = threadIdx.x >> 5;            // wave index == head
    const int lane = threadIdx.x & 31;
    const int half = lane >> 4;
    const int mr   = lane & 15;
    const int b    = blockIdx.x >> 7;             // SEQ/16 = 128 q-tiles per batch
    const int q0   = (blockIdx.x & 127) * 16;

    // qp A-fragments for this head: 16 rows x 64 K (two 32-K slabs), loaded once
    const _Float16* qrow = qp + ((size_t)(b * SEQ + q0 + mr)) * DM + h * HDIM;
    V16U aq0, aq1;
    aq0.h[0] = *(const v8h*)(qrow + 16 * half);
    aq0.h[1] = *(const v8h*)(qrow + 8 + 16 * half);
    aq1.h[0] = *(const v8h*)(qrow + 32 + 16 * half);
    aq1.h[1] = *(const v8h*)(qrow + 40 + 16 * half);

    v8f acc[4] = {};                               // ctx accumulator: 16 q x 64 d

    for (int k0 = 0; k0 < SEQ; k0 += 32) {
        // ---- scores: S_h = qp_tile (16x64) * kp_tile^T (64x32), two 16x16 n-tiles ----
        #pragma unroll
        for (int nt = 0; nt < 2; ++nt) {
            const int t = k0 + nt * 16 + mr;                 // token this lane supplies as B column
            const _Float16* krow = kp + ((size_t)(b * SEQ + t)) * DM + h * HDIM;
            V16U b0, b1;
            b0.v = *(const v16h*)(krow + 16 * half);         // K 0..31 slab
            b1.v = *(const v16h*)(krow + 32 + 16 * half);    // K 32..63 slab
            v8f s = {};
            s = __builtin_amdgcn_wmma_f32_16x16x32_f16(false, aq0.v, false, b0.v, (short)0, s, false, false);
            s = __builtin_amdgcn_wmma_f32_16x16x32_f16(false, aq1.v, false, b1.v, (short)0, s, false, false);
            #pragma unroll
            for (int r = 0; r < 8; ++r)
                S[h][r + 8 * half][nt * 16 + mr] = s[r];
        }
        __syncthreads();

        // ---- softmax across the 8 heads, elementwise over the 16x32 tile ----
        #pragma unroll
        for (int pp = 0; pp < 2; ++pp) {
            const int p  = threadIdx.x + pp * 256;
            const int sm = p >> 5, sn = p & 31;
            float mx = S[0][sm][sn];
            #pragma unroll
            for (int hh = 1; hh < NHEAD; ++hh) mx = fmaxf(mx, S[hh][sm][sn]);
            float e[NHEAD];
            float sum = 0.f;
            #pragma unroll
            for (int hh = 0; hh < NHEAD; ++hh) {
                e[hh] = __expf((S[hh][sm][sn] - mx) * 0.125f);   // 1/sqrt(64) folded in
                sum += e[hh];
            }
            const float inv = __builtin_amdgcn_rcpf(sum);
            #pragma unroll
            for (int hh = 0; hh < NHEAD; ++hh) S[hh][sm][sn] = e[hh] * inv;
        }
        __syncthreads();

        // ---- ctx += attn (16x32) * vp_slab (32x64): 4 output n-tiles ----
        V16U aat;   // attn A-fragment from LDS, contiguous K mapping
        #pragma unroll
        for (int j = 0; j < 8; ++j) aat.v[j]     = (_Float16)S[h][mr][j + 16 * half];
        #pragma unroll
        for (int j = 0; j < 8; ++j) aat.v[8 + j] = (_Float16)S[h][mr][8 + j + 16 * half];
        #pragma unroll
        for (int nt = 0; nt < 4; ++nt) {
            const int d = nt * 16 + mr;
            const _Float16* vcol = vpT + ((size_t)b * DM + h * HDIM + d) * SEQ + k0 + 16 * half;
            V16U bv;
            bv.v = *(const v16h*)vcol;
            acc[nt] = __builtin_amdgcn_wmma_f32_16x16x32_f16(false, aat.v, false, bv.v,
                                                             (short)0, acc[nt], false, false);
        }
        __syncthreads();                           // S reused next iteration

        if (k0 + 32 < SEQ)                         // warm L2/L0 for next kp slab
            __builtin_prefetch(kp + ((size_t)(b * SEQ + k0 + 32 + mr)) * DM + h * HDIM, 0, 1);
    }

    // ---- store ctx (f16, row-major [B*SEQ][DM]) ----
    #pragma unroll
    for (int nt = 0; nt < 4; ++nt) {
        #pragma unroll
        for (int r = 0; r < 8; ++r)
            ctx[((size_t)(b * SEQ + q0 + r + 8 * half)) * DM + h * HDIM + nt * 16 + mr] =
                (_Float16)acc[nt][r];
    }
}

// ---------------- launcher ----------------
extern "C" void kernel_launch(void* const* d_in, const int* in_sizes, int n_in,
                              void* d_out, int out_size, void* d_ws, size_t ws_size,
                              hipStream_t stream) {
    (void)in_sizes; (void)n_in; (void)out_size; (void)ws_size;
    const float* q  = (const float*)d_in[0];
    const float* k  = (const float*)d_in[1];
    const float* v  = (const float*)d_in[2];
    const float* Wq = (const float*)d_in[3];
    const float* bq = (const float*)d_in[4];
    const float* Wk = (const float*)d_in[5];
    const float* bk = (const float*)d_in[6];
    // d_in[7]=Wv, d_in[8]=bv: intentionally unused (reference bug preserved)
    const float* Wc = (const float*)d_in[9];
    const float* bc = (const float*)d_in[10];

    const size_t MT  = (size_t)BATCH * SEQ;   // 8192 tokens
    const size_t ACT = MT * DM;               // activation elements
    const size_t WSZ = (size_t)DM * DM;       // weight elements

    _Float16* w   = (_Float16*)d_ws;
    _Float16* qh  = w;
    _Float16* kh  = qh  + ACT;
    _Float16* vh  = kh  + ACT;
    _Float16* WqT = vh  + ACT;
    _Float16* WkT = WqT + WSZ;
    _Float16* WcT = WkT + WSZ;
    _Float16* qp  = WcT + WSZ;
    _Float16* kp  = qp  + ACT;
    _Float16* vpT = kp  + ACT;                // [B][DM][SEQ]
    _Float16* ctx = vpT + ACT;

    dim3 blk(256);
    cvt_f32_f16<<<dim3(ACT / 256), blk, 0, stream>>>(q, qh, (int)ACT);
    cvt_f32_f16<<<dim3(ACT / 256), blk, 0, stream>>>(k, kh, (int)ACT);
    cvt_f32_f16<<<dim3(ACT / 256), blk, 0, stream>>>(v, vh, (int)ACT);
    cvt_transpose_f32_f16<<<dim3(WSZ / 256), blk, 0, stream>>>(Wq, WqT);
    cvt_transpose_f32_f16<<<dim3(WSZ / 256), blk, 0, stream>>>(Wk, WkT);
    cvt_transpose_f32_f16<<<dim3(WSZ / 256), blk, 0, stream>>>(Wc, WcT);

    dim3 gg((unsigned)(MT / 16), (unsigned)(DM / 512));   // (512, 1)
    gemm_wmma<<<gg, blk, 0, stream>>>(qh, WqT, bq, qp,  (int)MT, DM, DM, 0);
    gemm_wmma<<<gg, blk, 0, stream>>>(kh, WkT, bk, kp,  (int)MT, DM, DM, 0);
    gemm_wmma<<<gg, blk, 0, stream>>>(vh, WkT, bk, vpT, (int)MT, DM, DM, 1); // vp uses Wk/bk (bug preserved)

    attn_kernel<<<dim3(BATCH * (SEQ / 16)), blk, 0, stream>>>(qp, kp, vpT, ctx);

    gemm_wmma<<<gg, blk, 0, stream>>>(ctx, WcT, bc, d_out, (int)MT, DM, DM, 2);
}
